// GraniteMoeHybridDecoderLayer_56650618635049
// MI455X (gfx1250) — matrix operand order repaired and checked by reading, not compiled, not run on hardware
//
#include <hip/hip_runtime.h>

typedef __bf16 bf16;
typedef __attribute__((ext_vector_type(16))) __bf16 v16bf;
typedef __attribute__((ext_vector_type(8)))  float  v8f;

#define HDIM   1536
#define NHEADS 12
#define NKVH   4
#define DHEAD  128
#define NEXP   8
#define IEXP   512
#define ISH    1024
#define SEQ    2048
#define BATCH  2
#define TTOK   (SEQ * BATCH)
#define SCALE_ATTN 0.0078125f
#define RES_MULT   0.22f
#define RMS_EPS    1e-6f

// ---------------------------------------------------------------------------
// WMMA helpers (CDNA5 wave32, 16x16x32 bf16 -> f32)
// ---------------------------------------------------------------------------
__device__ __forceinline__ v8f wmma_bf16(v16bf a, v16bf b, v8f c) {
  return __builtin_amdgcn_wmma_f32_16x16x32_bf16(
      /*neg_a=*/false, a, /*neg_b=*/false, b,
      /*c_mod=*/(short)0, c, /*reuse_a=*/false, /*reuse_b=*/false);
}

// Async global -> LDS copy (16 bytes / lane), tracked by ASYNCcnt.
__device__ __forceinline__ void async_load_b128(unsigned lds_addr, const void* gaddr) {
  asm volatile("global_load_async_to_lds_b128 %0, %1, off"
               :: "v"(lds_addr), "v"(gaddr) : "memory");
}
__device__ __forceinline__ void wait_async0() {
  asm volatile("s_wait_asynccnt 0x0" ::: "memory");
}
// Low 32 bits of a generic pointer to LDS == LDS byte address (aperture rule).
__device__ __forceinline__ unsigned lds_addr_of(const void* p) {
  return (unsigned)(uintptr_t)p;
}

// A fragment: 16 rows x 32 K, row-major source with leading dim `ld` (bf16).
__device__ __forceinline__ v16bf load_a_frag(const bf16* base, int ld) {
  int lane = threadIdx.x & 31;
  int row  = lane & 15;
  int kb   = (lane >> 4) * 16;
  const bf16* p = base + row * ld + kb;
  v16bf a;
#pragma unroll
  for (int j = 0; j < 16; ++j) a[j] = p[j];
  return a;
}

// B fragment from weight-style layout W[n][k] (row-major, leading dim ld).
__device__ __forceinline__ v16bf load_b_frag(const bf16* base, int ld) {
  int lane = threadIdx.x & 31;
  int col  = lane & 15;
  int kb   = (lane >> 4) * 16;
  const bf16* p = base + col * ld + kb;
  v16bf b;
#pragma unroll
  for (int j = 0; j < 16; ++j) b[j] = p[j];
  return b;
}

// B fragment from true row-major B[k][n] (leading dim ld) — used for P*V.
__device__ __forceinline__ v16bf load_bT_frag(const bf16* base, int ld) {
  int lane = threadIdx.x & 31;
  int col  = lane & 15;
  int kb   = (lane >> 4) * 16;
  v16bf b;
#pragma unroll
  for (int j = 0; j < 16; ++j) b[j] = base[(kb + j) * ld + col];
  return b;
}

// ---------------------------------------------------------------------------
// Generic bf16 WMMA GEMM:  C[M,N] = A[M,K] * W[N,K]^T
// block tile 128(M) x 128(N), 8 waves (4x2), wave tile 32x64 (2x4 frags).
// Double-buffered LDS fed by async global->LDS copies.
// mode 0: outF = acc
// mode 1: outB = bf16(acc)
// mode 2: outF += (rowscale ? rowscale[m*rs_stride] : 1) * acc
// mode 3: outF = res + alpha*acc
// ---------------------------------------------------------------------------
__global__ void __launch_bounds__(256)
gemm_bf16_wmma(const bf16* __restrict__ A, const bf16* __restrict__ W,
               int M, int N, int K,
               float* __restrict__ outF, bf16* __restrict__ outB,
               const float* __restrict__ res, float alpha,
               const float* __restrict__ rowscale, int rs_stride, int mode) {
  __shared__ bf16 As[2][128 * 32];
  __shared__ bf16 Ws[2][128 * 32];
  const int tid  = threadIdx.x;
  const int wave = tid >> 5;
  const int m0 = blockIdx.y * 128;
  const int n0 = blockIdx.x * 128;
  const int wm = (wave >> 1) * 32;   // 0,32,64,96
  const int wn = (wave & 1) * 64;    // 0,64

  v8f acc[2][4];
#pragma unroll
  for (int i = 0; i < 2; ++i)
#pragma unroll
    for (int j = 0; j < 4; ++j) acc[i][j] = (v8f){};

  // per-thread staging: one 16-elem (32B) segment of a 128x32 tile
  const int sr = tid >> 1, sc = (tid & 1) * 16;
  const bf16* gA = A + (size_t)(m0 + sr) * K + sc;
  const bf16* gW = W + (size_t)(n0 + sr) * K + sc;
  const unsigned laA0 = lds_addr_of(&As[0][sr * 32 + sc]);
  const unsigned laW0 = lds_addr_of(&Ws[0][sr * 32 + sc]);
  const unsigned bufStride = (unsigned)(128 * 32 * sizeof(bf16));

  const int nk = K / 32;
  // prologue: issue tile 0
  async_load_b128(laA0,      gA);
  async_load_b128(laA0 + 16, gA + 8);
  async_load_b128(laW0,      gW);
  async_load_b128(laW0 + 16, gW + 8);

  for (int i = 0; i < nk; ++i) {
    wait_async0();
    __syncthreads();
    if (i + 1 < nk) {  // issue next tile into the other buffer
      const bf16* a = gA + (size_t)(i + 1) * 32;
      const bf16* w = gW + (size_t)(i + 1) * 32;
      unsigned off = ((i + 1) & 1) * bufStride;
      async_load_b128(laA0 + off,      a);
      async_load_b128(laA0 + off + 16, a + 8);
      async_load_b128(laW0 + off,      w);
      async_load_b128(laW0 + off + 16, w + 8);
    }
    const int buf = i & 1;
    v16bf a0 = load_a_frag(&As[buf][(wm + 0)  * 32], 32);
    v16bf a1 = load_a_frag(&As[buf][(wm + 16) * 32], 32);
    v16bf b0 = load_b_frag(&Ws[buf][(wn + 0)  * 32], 32);
    v16bf b1 = load_b_frag(&Ws[buf][(wn + 16) * 32], 32);
    v16bf b2 = load_b_frag(&Ws[buf][(wn + 32) * 32], 32);
    v16bf b3 = load_b_frag(&Ws[buf][(wn + 48) * 32], 32);
    acc[0][0] = wmma_bf16(a0, b0, acc[0][0]);
    acc[0][1] = wmma_bf16(a0, b1, acc[0][1]);
    acc[0][2] = wmma_bf16(a0, b2, acc[0][2]);
    acc[0][3] = wmma_bf16(a0, b3, acc[0][3]);
    acc[1][0] = wmma_bf16(a1, b0, acc[1][0]);
    acc[1][1] = wmma_bf16(a1, b1, acc[1][1]);
    acc[1][2] = wmma_bf16(a1, b2, acc[1][2]);
    acc[1][3] = wmma_bf16(a1, b3, acc[1][3]);
    __syncthreads();
  }

  const int lane = tid & 31;
  const int lrow = (lane >> 4) * 8;
  const int lcol = lane & 15;
#pragma unroll
  for (int fm = 0; fm < 2; ++fm)
#pragma unroll
    for (int fn = 0; fn < 4; ++fn)
#pragma unroll
      for (int r = 0; r < 8; ++r) {
        int gm = m0 + wm + fm * 16 + lrow + r;
        int gn = n0 + wn + fn * 16 + lcol;
        size_t idx = (size_t)gm * N + gn;
        float v = acc[fm][fn][r];
        if (mode == 0) {
          outF[idx] = v;
        } else if (mode == 1) {
          outB[idx] = (bf16)v;
        } else if (mode == 2) {
          float sc2 = rowscale ? rowscale[(size_t)gm * rs_stride] : 1.0f;
          outF[idx] += sc2 * v;
        } else {
          outF[idx] = res[idx] + alpha * v;
        }
      }
}

// ---------------------------------------------------------------------------
// RMSNorm (fp32 in, bf16 out). One block (256 thr) per token.
// ---------------------------------------------------------------------------
__global__ void __launch_bounds__(256)
rmsnorm_kernel(const float* __restrict__ x, const float* __restrict__ w,
               bf16* __restrict__ out, int H) {
  __shared__ float red[8];
  const size_t t = blockIdx.x;
  const float* xr = x + t * H;
  float ss = 0.f;
  for (int i = threadIdx.x; i < H; i += 256) { float v = xr[i]; ss += v * v; }
#pragma unroll
  for (int off = 16; off; off >>= 1) ss += __shfl_down(ss, off, 32);
  if ((threadIdx.x & 31) == 0) red[threadIdx.x >> 5] = ss;
  __syncthreads();
  if (threadIdx.x == 0) {
    float tot = 0.f;
#pragma unroll
    for (int i = 0; i < 8; ++i) tot += red[i];
    red[0] = tot;
  }
  __syncthreads();
  float inv = rsqrtf(red[0] / (float)H + RMS_EPS);
  for (int i = threadIdx.x; i < H; i += 256)
    out[t * H + i] = (bf16)(w[i] * xr[i] * inv);
}

// ---------------------------------------------------------------------------
// RoPE (fp32 in, bf16 out). layout [T, nheads, 128].
// ---------------------------------------------------------------------------
__global__ void __launch_bounds__(256)
rope_kernel(const float* __restrict__ qk, const float* __restrict__ cosT,
            const float* __restrict__ sinT, const int* __restrict__ pos,
            bf16* __restrict__ out, int nheads, size_t total) {
  size_t idx = (size_t)blockIdx.x * blockDim.x + threadIdx.x;
  if (idx >= total) return;
  int d = (int)(idx & 127);
  size_t th = idx >> 7;
  int t = (int)(th / (size_t)nheads);
  float x = qk[idx];
  float other = (d < 64) ? -qk[idx + 64] : qk[idx - 64];
  int p = pos[t];
  float c = cosT[(size_t)p * DHEAD + d];
  float s = sinT[(size_t)p * DHEAD + d];
  out[idx] = (bf16)(x * c + other * s);
}

// ---------------------------------------------------------------------------
// Flash attention (causal, GQA). Block = 4 waves, 64 q-rows of one (b,h).
// grid: (SEQ/64, NHEADS, BATCH). K/V/Q staged with async global->LDS.
// ---------------------------------------------------------------------------
__global__ void __launch_bounds__(128)
flash_attn_kernel(const bf16* __restrict__ Q, const bf16* __restrict__ Kh,
                  const bf16* __restrict__ Vh, bf16* __restrict__ O) {
  __shared__ bf16  Qs[64 * 128];
  __shared__ bf16  Ks[32 * 128];
  __shared__ bf16  Vs[32 * 128];
  __shared__ float Sb[64 * 32];
  __shared__ bf16  Pb[64 * 32];
  __shared__ float m_i[64], l_i[64], fac[64];

  const int tid  = threadIdx.x;
  const int wave = tid >> 5;
  const int lane = tid & 31;
  const int q0 = blockIdx.x * 64;
  const int h  = blockIdx.y;
  const int b  = blockIdx.z;
  const int hk = h / (NHEADS / NKVH);

  const size_t qbase = ((size_t)b * SEQ + q0) * (NHEADS * DHEAD) + (size_t)h * DHEAD;
  for (int i = tid; i < 64 * 8; i += 128) {  // 64 rows x 8 segs of 16 bf16
    int r = i >> 3, c = (i & 7) * 16;
    const bf16* g = Q + qbase + (size_t)r * NHEADS * DHEAD + c;
    unsigned l = lds_addr_of(&Qs[r * 128 + c]);
    async_load_b128(l, g);
    async_load_b128(l + 16, g + 8);
  }
  if (tid < 64) { m_i[tid] = -1e30f; l_i[tid] = 0.f; }
  wait_async0();
  __syncthreads();

  v8f o_acc[8];
#pragma unroll
  for (int i = 0; i < 8; ++i) o_acc[i] = (v8f){};

  const int lrow = (lane >> 4) * 8;
  const int lcol = lane & 15;

  for (int j0 = 0; j0 <= q0 + 63; j0 += 32) {
    const size_t kvbase = ((size_t)b * SEQ + j0) * (NKVH * DHEAD) + (size_t)hk * DHEAD;
    for (int i = tid; i < 32 * 8; i += 128) {
      int r = i >> 3, c = (i & 7) * 16;
      const bf16* gk = Kh + kvbase + (size_t)r * NKVH * DHEAD + c;
      const bf16* gv = Vh + kvbase + (size_t)r * NKVH * DHEAD + c;
      unsigned lk = lds_addr_of(&Ks[r * 128 + c]);
      unsigned lv = lds_addr_of(&Vs[r * 128 + c]);
      async_load_b128(lk, gk);
      async_load_b128(lk + 16, gk + 8);
      async_load_b128(lv, gv);
      async_load_b128(lv + 16, gv + 8);
    }
    wait_async0();
    __syncthreads();

    // S = Q*K^T for this wave's 16 rows, 32 keys (2 frags), K-dim = 128
    v8f s0 = (v8f){}, s1 = (v8f){};
#pragma unroll
    for (int kk = 0; kk < 128; kk += 32) {
      v16bf aq  = load_a_frag(&Qs[(wave * 16) * 128 + kk], 128);
      v16bf bk0 = load_b_frag(&Ks[0 * 128 + kk], 128);
      v16bf bk1 = load_b_frag(&Ks[16 * 128 + kk], 128);
      s0 = wmma_bf16(aq, bk0, s0);
      s1 = wmma_bf16(aq, bk1, s1);
    }
#pragma unroll
    for (int r = 0; r < 8; ++r) {
      int row = wave * 16 + lrow + r;
      float v0 = s0[r] * SCALE_ATTN;
      float v1 = s1[r] * SCALE_ATTN;
      if (j0 + lcol      > q0 + row) v0 = -1e30f;
      if (j0 + 16 + lcol > q0 + row) v1 = -1e30f;
      Sb[row * 32 + lcol]      = v0;
      Sb[row * 32 + 16 + lcol] = v1;
    }
    __syncthreads();

    // online softmax per row (threads 0..63)
    if (tid < 64) {
      int row = tid;
      float mOld = m_i[row];
      float mt = mOld;
      for (int c = 0; c < 32; ++c) mt = fmaxf(mt, Sb[row * 32 + c]);
      float sc = __expf(mOld - mt);
      float rs = 0.f;
      for (int c = 0; c < 32; ++c) {
        float sv = Sb[row * 32 + c];
        float p = (sv < -1e29f) ? 0.f : __expf(sv - mt);
        Pb[row * 32 + c] = (bf16)p;
        rs += p;
      }
      m_i[row] = mt;
      l_i[row] = l_i[row] * sc + rs;
      fac[row] = sc;
    }
    __syncthreads();

    // rescale O, then O += P*V  (P: 16x32, V: 32x128 row-major)
    float f[8];
#pragma unroll
    for (int r = 0; r < 8; ++r) f[r] = fac[wave * 16 + lrow + r];
    v16bf ap = load_a_frag(&Pb[(wave * 16) * 32], 32);
#pragma unroll
    for (int d0 = 0; d0 < 8; ++d0) {
#pragma unroll
      for (int r = 0; r < 8; ++r) o_acc[d0][r] *= f[r];
      v16bf bv = load_bT_frag(&Vs[d0 * 16], 128);
      o_acc[d0] = wmma_bf16(ap, bv, o_acc[d0]);
    }
    __syncthreads();
  }

  // epilogue: divide by l, store bf16 [B,S,NH*D]
#pragma unroll
  for (int d0 = 0; d0 < 8; ++d0)
#pragma unroll
    for (int r = 0; r < 8; ++r) {
      int row = wave * 16 + lrow + r;
      float invl = 1.f / l_i[row];
      size_t oidx = ((size_t)b * SEQ + q0 + row) * (NHEADS * DHEAD)
                  + (size_t)h * DHEAD + d0 * 16 + lcol;
      O[oidx] = (bf16)(o_acc[d0][r] * invl);
    }
}

// ---------------------------------------------------------------------------
// Router: one wave per token; E=8 logits, top-2 softmax -> dense combine[T,8]
// ---------------------------------------------------------------------------
__global__ void __launch_bounds__(256)
router_kernel(const bf16* __restrict__ xn, const float* __restrict__ rw,
              float* __restrict__ combine, int H) {
  const int wave = threadIdx.x >> 5, lane = threadIdx.x & 31;
  const int t = blockIdx.x * 8 + wave;
  const bf16* xr = xn + (size_t)t * H;
  float logits[NEXP];
#pragma unroll
  for (int e = 0; e < NEXP; ++e) {
    const float* wr = rw + (size_t)e * H;
    float acc = 0.f;
    for (int i = lane; i < H; i += 32) acc += (float)xr[i] * wr[i];
#pragma unroll
    for (int off = 16; off; off >>= 1) acc += __shfl_down(acc, off, 32);
    logits[e] = __shfl(acc, 0, 32);
  }
  if (lane == 0) {
    int i1 = 0;
    for (int e = 1; e < NEXP; ++e) if (logits[e] > logits[i1]) i1 = e;
    int i2 = -1;
    for (int e = 0; e < NEXP; ++e)
      if (e != i1 && (i2 < 0 || logits[e] > logits[i2])) i2 = e;
    float e1 = 1.0f;  // exp(0) relative to max
    float e2 = __expf(logits[i2] - logits[i1]);
    float s = e1 + e2;
    float outv[NEXP];
#pragma unroll
    for (int e = 0; e < NEXP; ++e) outv[e] = 0.f;
    outv[i1] = e1 / s; outv[i2] = e2 / s;
#pragma unroll
    for (int e = 0; e < NEXP; ++e) combine[(size_t)t * NEXP + e] = outv[e];
  }
}

// ---------------------------------------------------------------------------
// Elementwise kernels
// ---------------------------------------------------------------------------
__global__ void __launch_bounds__(256)
cast_bf16_kernel(const float* __restrict__ in, bf16* __restrict__ out, size_t n) {
  for (size_t i = (size_t)blockIdx.x * blockDim.x + threadIdx.x; i < n;
       i += (size_t)gridDim.x * blockDim.x)
    out[i] = (bf16)in[i];
}

__global__ void __launch_bounds__(256)
silu_mul_kernel(const float* __restrict__ g, const float* __restrict__ u,
                bf16* __restrict__ out, size_t n) {
  for (size_t i = (size_t)blockIdx.x * blockDim.x + threadIdx.x; i < n;
       i += (size_t)gridDim.x * blockDim.x) {
    float gv = g[i];
    float sg = gv / (1.f + __expf(-gv));
    out[i] = (bf16)(sg * u[i]);
  }
}

__global__ void __launch_bounds__(256)
silu_mul_split_kernel(const float* __restrict__ y, bf16* __restrict__ out,
                      int I, int ld, size_t n) {
  for (size_t idx = (size_t)blockIdx.x * blockDim.x + threadIdx.x; idx < n;
       idx += (size_t)gridDim.x * blockDim.x) {
    size_t t = idx / (size_t)I;
    int i = (int)(idx - t * I);
    float a = y[t * ld + i];
    float bg = y[t * ld + I + i];
    float sg = a / (1.f + __expf(-a));
    out[idx] = (bf16)(sg * bg);
  }
}

__global__ void __launch_bounds__(256)
final_add_kernel(const float* __restrict__ h1, const float* __restrict__ ffn,
                 float* __restrict__ out, size_t n) {
  for (size_t i = (size_t)blockIdx.x * blockDim.x + threadIdx.x; i < n;
       i += (size_t)gridDim.x * blockDim.x)
    out[i] = h1[i] + ffn[i] * RES_MULT;
}

// ---------------------------------------------------------------------------
// Host orchestration
// ---------------------------------------------------------------------------
extern "C" void kernel_launch(void* const* d_in, const int* in_sizes, int n_in,
                              void* d_out, int out_size, void* d_ws, size_t ws_size,
                              hipStream_t stream) {
  const float* hs       = (const float*)d_in[0];
  const int*   pos      = (const int*)d_in[1];
  const float* cosT     = (const float*)d_in[2];
  const float* sinT     = (const float*)d_in[3];
  const float* ln1      = (const float*)d_in[4];
  const float* ln2      = (const float*)d_in[5];
  const float* wq       = (const float*)d_in[6];
  const float* wk       = (const float*)d_in[7];
  const float* wv       = (const float*)d_in[8];
  const float* wo       = (const float*)d_in[9];
  const float* router_w = (const float*)d_in[10];
  const float* w_gate   = (const float*)d_in[11];
  const float* w_up     = (const float*)d_in[12];
  const float* w_down   = (const float*)d_in[13];
  const float* sh_in    = (const float*)d_in[14];
  const float* sh_out   = (const float*)d_in[15];
  float* out = (float*)d_out;

  char* wp = (char*)d_ws;
  auto alloc = [&](size_t bytes) -> void* {
    void* r = (void*)wp;
    wp += (bytes + 255) & ~(size_t)255;
    return r;
  };

  const size_t T = TTOK, H = HDIM;
  bf16* xn1     = (bf16*)alloc(T * H * 2);
  bf16* wq_b    = (bf16*)alloc((size_t)H * H * 2);
  bf16* wk_b    = (bf16*)alloc((size_t)NKVH * DHEAD * H * 2);
  bf16* wv_b    = (bf16*)alloc((size_t)NKVH * DHEAD * H * 2);
  bf16* wo_b    = (bf16*)alloc((size_t)H * H * 2);
  bf16* wg_b    = (bf16*)alloc((size_t)NEXP * IEXP * H * 2);
  bf16* wu_b    = (bf16*)alloc((size_t)NEXP * IEXP * H * 2);
  bf16* wd_b    = (bf16*)alloc((size_t)NEXP * H * IEXP * 2);
  bf16* shin_b  = (bf16*)alloc((size_t)2 * ISH * H * 2);
  bf16* shout_b = (bf16*)alloc((size_t)H * ISH * 2);
  float* qf     = (float*)alloc(T * (size_t)NHEADS * DHEAD * 4);
  float* kf     = (float*)alloc(T * (size_t)NKVH * DHEAD * 4);
  bf16* q_b     = (bf16*)alloc(T * (size_t)NHEADS * DHEAD * 2);
  bf16* k_b     = (bf16*)alloc(T * (size_t)NKVH * DHEAD * 2);
  bf16* v_b     = (bf16*)alloc(T * (size_t)NKVH * DHEAD * 2);
  bf16* attn_b  = (bf16*)alloc(T * H * 2);
  float* h1     = (float*)alloc(T * H * 4);
  bf16* xn2     = (bf16*)alloc(T * H * 2);
  float* comb   = (float*)alloc(T * NEXP * 4);
  float* moe    = (float*)alloc(T * H * 4);
  float* gbuf   = (float*)alloc(T * (size_t)IEXP * 4);
  float* ubuf   = (float*)alloc(T * (size_t)IEXP * 4);
  bf16* actb    = (bf16*)alloc(T * (size_t)IEXP * 2);
  float* ybuf   = (float*)alloc(T * (size_t)(2 * ISH) * 4);
  bf16* sact    = (bf16*)alloc(T * (size_t)ISH * 2);

  const dim3 blk256(256), blk128(128);
  auto cast = [&](const float* s, bf16* d, size_t n) {
    int g = (int)((n + 256 * 4 - 1) / (256 * 4));
    if (g > 4096) g = 4096;
    cast_bf16_kernel<<<g, blk256, 0, stream>>>(s, d, n);
  };
  cast(wq, wq_b, (size_t)H * H);
  cast(wk, wk_b, (size_t)NKVH * DHEAD * H);
  cast(wv, wv_b, (size_t)NKVH * DHEAD * H);
  cast(wo, wo_b, (size_t)H * H);
  cast(w_gate, wg_b, (size_t)NEXP * IEXP * H);
  cast(w_up,   wu_b, (size_t)NEXP * IEXP * H);
  cast(w_down, wd_b, (size_t)NEXP * H * IEXP);
  cast(sh_in,  shin_b, (size_t)2 * ISH * H);
  cast(sh_out, shout_b, (size_t)H * ISH);

  // ---- attention block ----
  rmsnorm_kernel<<<(int)T, blk256, 0, stream>>>(hs, ln1, xn1, (int)H);

  gemm_bf16_wmma<<<dim3(NHEADS * DHEAD / 128, (int)T / 128), blk256, 0, stream>>>(
      xn1, wq_b, (int)T, NHEADS * DHEAD, (int)H, qf, nullptr, nullptr, 0.f, nullptr, 0, 0);
  gemm_bf16_wmma<<<dim3(NKVH * DHEAD / 128, (int)T / 128), blk256, 0, stream>>>(
      xn1, wk_b, (int)T, NKVH * DHEAD, (int)H, kf, nullptr, nullptr, 0.f, nullptr, 0, 0);
  gemm_bf16_wmma<<<dim3(NKVH * DHEAD / 128, (int)T / 128), blk256, 0, stream>>>(
      xn1, wv_b, (int)T, NKVH * DHEAD, (int)H, nullptr, v_b, nullptr, 0.f, nullptr, 0, 1);

  {
    size_t nq = T * (size_t)NHEADS * DHEAD;
    rope_kernel<<<(int)((nq + 255) / 256), blk256, 0, stream>>>(qf, cosT, sinT, pos, q_b, NHEADS, nq);
    size_t nk = T * (size_t)NKVH * DHEAD;
    rope_kernel<<<(int)((nk + 255) / 256), blk256, 0, stream>>>(kf, cosT, sinT, pos, k_b, NKVH, nk);
  }

  flash_attn_kernel<<<dim3(SEQ / 64, NHEADS, BATCH), blk128, 0, stream>>>(q_b, k_b, v_b, attn_b);

  gemm_bf16_wmma<<<dim3((int)H / 128, (int)T / 128), blk256, 0, stream>>>(
      attn_b, wo_b, (int)T, (int)H, (int)H, h1, nullptr, hs, RES_MULT, nullptr, 0, 3);

  // ---- MoE + shared MLP block ----
  rmsnorm_kernel<<<(int)T, blk256, 0, stream>>>(h1, ln2, xn2, (int)H);
  router_kernel<<<(int)T / 8, blk256, 0, stream>>>(xn2, router_w, comb, (int)H);
  hipMemsetAsync(moe, 0, T * H * 4, stream);

  for (int e = 0; e < NEXP; ++e) {
    const bf16* wg_e = wg_b + (size_t)e * IEXP * H;
    const bf16* wu_e = wu_b + (size_t)e * IEXP * H;
    const bf16* wd_e = wd_b + (size_t)e * H * IEXP;
    gemm_bf16_wmma<<<dim3(IEXP / 128, (int)T / 128), blk256, 0, stream>>>(
        xn2, wg_e, (int)T, IEXP, (int)H, gbuf, nullptr, nullptr, 0.f, nullptr, 0, 0);
    gemm_bf16_wmma<<<dim3(IEXP / 128, (int)T / 128), blk256, 0, stream>>>(
        xn2, wu_e, (int)T, IEXP, (int)H, ubuf, nullptr, nullptr, 0.f, nullptr, 0, 0);
    size_t na = T * (size_t)IEXP;
    silu_mul_kernel<<<4096, blk256, 0, stream>>>(gbuf, ubuf, actb, na);
    gemm_bf16_wmma<<<dim3((int)H / 128, (int)T / 128), blk256, 0, stream>>>(
        actb, wd_e, (int)T, (int)H, IEXP, moe, nullptr, nullptr, 0.f, comb + e, NEXP, 2);
  }

  gemm_bf16_wmma<<<dim3(2 * ISH / 128, (int)T / 128), blk256, 0, stream>>>(
      xn2, shin_b, (int)T, 2 * ISH, (int)H, ybuf, nullptr, nullptr, 0.f, nullptr, 0, 0);
  {
    size_t ns = T * (size_t)ISH;
    silu_mul_split_kernel<<<4096, blk256, 0, stream>>>(ybuf, sact, ISH, 2 * ISH, ns);
  }
  gemm_bf16_wmma<<<dim3((int)H / 128, (int)T / 128), blk256, 0, stream>>>(
      sact, shout_b, (int)T, (int)H, ISH, moe, nullptr, nullptr, 0.f, nullptr, 0, 2);

  final_add_kernel<<<4096, blk256, 0, stream>>>(h1, moe, out, T * H);
}
